// SlotAttention_84095459656318
// MI455X (gfx1250) — matrix operand order repaired and checked by reading, not compiled, not run on hardware
//
#include <hip/hip_runtime.h>

// Problem constants
#define BB   64
#define NN   4096
#define DD   256
#define NSL  8
#define NITR 3

typedef __attribute__((ext_vector_type(16))) __bf16 bf16x16;
typedef __attribute__((ext_vector_type(8)))  float  f32x8;

__device__ __forceinline__ unsigned short f2bf(float f) {
  unsigned u = __builtin_bit_cast(unsigned, f);
  u += 0x7FFFu + ((u >> 16) & 1u);            // round-to-nearest-even
  return (unsigned short)(u >> 16);
}
__device__ __forceinline__ float bf2f(unsigned short h) {
  return __builtin_bit_cast(float, ((unsigned)h) << 16);
}
// A/B fragment loader: two 16-byte chunks (works for global and LDS pointers)
__device__ __forceinline__ bf16x16 ldfrag(const unsigned short* p, int o0, int o1) {
  union { bf16x16 v; uint4 q[2]; } f;
  f.q[0] = *reinterpret_cast<const uint4*>(p + o0);
  f.q[1] = *reinterpret_cast<const uint4*>(p + o1);
  return f.v;
}
__device__ __forceinline__ f32x8 wmma_bf(bf16x16 a, bf16x16 b, f32x8 c) {
  return __builtin_amdgcn_wmma_f32_16x16x32_bf16(false, a, false, b, (short)0, c, false, false);
}

// ---------------------------------------------------------------------------
// f32 -> bf16 weight conversion
__global__ void k_conv(const float* __restrict__ s, unsigned short* __restrict__ d, int n) {
  int i = blockIdx.x * blockDim.x + threadIdx.x;
  if (i < n) d[i] = f2bf(s[i]);
}

// slots = mu + exp(log_sigma) * noise
__global__ void k_slots_init(const float* __restrict__ noise, const float* __restrict__ mu,
                             const float* __restrict__ ls, float* __restrict__ slots) {
  int i = blockIdx.x * blockDim.x + threadIdx.x;
  if (i < BB * NSL * DD) {
    int d = i & (DD - 1);
    slots[i] = mu[d] + __expf(ls[d]) * noise[i];
  }
}

// ---------------------------------------------------------------------------
// Fused LN(inputs) + K/V projection.  32 input rows per WG (8 waves).
// k stored row-major bf16 [B*N, D]; v stored transposed bf16 per batch [D, N].
__global__ void __launch_bounds__(256) k_lnkv(
    const float* __restrict__ inp, const float* __restrict__ g_in, const float* __restrict__ be_in,
    const unsigned short* __restrict__ Wkb, const float* __restrict__ bk,
    const unsigned short* __restrict__ Wvb, const float* __restrict__ bv,
    unsigned short* __restrict__ kbf, unsigned short* __restrict__ vT) {
  __shared__ unsigned short X[32 * 264];      // padded stride breaks bank alignment
  const int lane = threadIdx.x & 31, wave = threadIdx.x >> 5;
  const int blk = blockIdx.x;

  // LayerNorm: each wave normalizes 4 rows
  for (int rr = 0; rr < 4; ++rr) {
    int r = wave * 4 + rr;
    const float* row = inp + ((long)blk * 32 + r) * DD;
    float x[8], s1 = 0.f, s2 = 0.f;
#pragma unroll
    for (int e = 0; e < 8; ++e) { x[e] = row[lane + 32 * e]; s1 += x[e]; s2 += x[e] * x[e]; }
#pragma unroll
    for (int m = 16; m >= 1; m >>= 1) { s1 += __shfl_xor(s1, m, 32); s2 += __shfl_xor(s2, m, 32); }
    float mean = s1 * (1.f / DD);
    float rstd = rsqrtf(s2 * (1.f / DD) - mean * mean + 1e-5f);
#pragma unroll
    for (int e = 0; e < 8; ++e) {
      int d = lane + 32 * e;
      X[r * 264 + d] = f2bf((x[e] - mean) * rstd * g_in[d] + be_in[d]);
    }
  }
  __syncthreads();

  const int cn = lane & 15, hf = lane >> 4;
  for (int s = 0; s < 8; ++s) {
    int tt = s * 8 + wave;                    // 64 tiles: 2 mats x 2 rowtiles x 16 coltiles
    int mat = tt >> 5, rt = (tt >> 4) & 1, ct = tt & 15;
    const unsigned short* W = mat ? Wvb : Wkb;
    const unsigned short* Arow = X + (rt * 16 + cn) * 264;
    const unsigned short* Brow = W + (ct * 16 + cn) * DD;
    f32x8 acc = {0.f, 0.f, 0.f, 0.f, 0.f, 0.f, 0.f, 0.f};
#pragma unroll
    for (int kk = 0; kk < 8; ++kk) {
      int k0 = kk * 32;
      bf16x16 a = ldfrag(Arow, k0 + hf * 16, k0 + hf * 16 + 8);
      bf16x16 b = ldfrag(Brow, k0 + hf * 16, k0 + hf * 16 + 8);
      acc = wmma_bf(a, b, acc);
    }
    int col = ct * 16 + cn;
    float bias = mat ? bv[col] : bk[col];
    if (mat == 0) {                           // k: row-major, scalar b16 stores
      long jbase = (long)blk * 32 + rt * 16 + hf * 8;
#pragma unroll
      for (int g = 0; g < 8; ++g)
        kbf[(jbase + g) * DD + col] = f2bf(acc[g] + bias);
    } else {                                  // v: transposed, 8 consecutive j -> one b128 store
      long j = (long)blk * 32 + rt * 16 + hf * 8;
      int bat = (int)(j >> 12), jj = (int)(j & (NN - 1));
      unsigned short h8[8];
#pragma unroll
      for (int g = 0; g < 8; ++g) h8[g] = f2bf(acc[g] + bias);
      uint4 pk;
      pk.x = (unsigned)h8[0] | ((unsigned)h8[1] << 16);
      pk.y = (unsigned)h8[2] | ((unsigned)h8[3] << 16);
      pk.z = (unsigned)h8[4] | ((unsigned)h8[5] << 16);
      pk.w = (unsigned)h8[6] | ((unsigned)h8[7] << 16);
      *reinterpret_cast<uint4*>(vT + ((long)(bat * DD + col)) * NN + jj) = pk;
    }
  }
}

// ---------------------------------------------------------------------------
// Per-iteration: q = LN(slots) @ Wq^T + bq.  One WG per 2 batches (16 slot rows).
// Also emits slots_bf16 (raw, zero-padded rows 8..15) for the GRU h-path.
__global__ void __launch_bounds__(256) k_qproj(
    const float* __restrict__ slots, const float* __restrict__ g_sl, const float* __restrict__ be_sl,
    const unsigned short* __restrict__ Wqb, const float* __restrict__ bq,
    unsigned short* __restrict__ qbf, unsigned short* __restrict__ slotsbf) {
  __shared__ unsigned short A[16 * 264];
  const int lane = threadIdx.x & 31, wave = threadIdx.x >> 5;
  const int b0 = blockIdx.x * 2;

  for (int rr = 0; rr < 2; ++rr) {
    int r = wave * 2 + rr;                    // 0..15
    int bat = b0 + (r >> 3), sl = r & 7;
    const float* row = slots + ((long)bat * NSL + sl) * DD;
    float x[8], s1 = 0.f, s2 = 0.f;
#pragma unroll
    for (int e = 0; e < 8; ++e) { x[e] = row[lane + 32 * e]; s1 += x[e]; s2 += x[e] * x[e]; }
#pragma unroll
    for (int m = 16; m >= 1; m >>= 1) { s1 += __shfl_xor(s1, m, 32); s2 += __shfl_xor(s2, m, 32); }
    float mean = s1 * (1.f / DD);
    float rstd = rsqrtf(s2 * (1.f / DD) - mean * mean + 1e-5f);
#pragma unroll
    for (int e = 0; e < 8; ++e) {
      int d = lane + 32 * e;
      A[r * 264 + d] = f2bf((x[e] - mean) * rstd * g_sl[d] + be_sl[d]);
      slotsbf[((long)bat * 16 + sl) * DD + d] = f2bf(x[e]);
      slotsbf[((long)bat * 16 + 8 + sl) * DD + d] = 0;
    }
  }
  __syncthreads();

  const int cn = lane & 15, hf = lane >> 4;
  for (int s = 0; s < 2; ++s) {
    int ct = wave * 2 + s;
    const unsigned short* Arow = A + cn * 264;
    const unsigned short* Brow = Wqb + (ct * 16 + cn) * DD;
    f32x8 acc = {0.f, 0.f, 0.f, 0.f, 0.f, 0.f, 0.f, 0.f};
#pragma unroll
    for (int kk = 0; kk < 8; ++kk) {
      int k0 = kk * 32;
      acc = wmma_bf(ldfrag(Arow, k0 + hf * 16, k0 + hf * 16 + 8),
                    ldfrag(Brow, k0 + hf * 16, k0 + hf * 16 + 8), acc);
    }
    int col = ct * 16 + cn;
    float bias = bq[col];
#pragma unroll
    for (int g = 0; g < 8; ++g) {
      int r = g + 8 * hf;
      int bat = b0 + (r >> 3), sl = r & 7;
      qbf[((long)bat * 16 + sl) * DD + col] = f2bf(acc[g] + bias);
      qbf[((long)bat * 16 + 8 + sl) * DD + col] = 0;
    }
  }
}

// ---------------------------------------------------------------------------
// dots = q @ k^T * SCALE; softmax over slots (lane-local: lanes 0..15 hold all
// 8 slot rows of one column); store attn(+EPS) bf16, rows 8..15 zeroed.
__global__ void __launch_bounds__(256) k_dots(
    const unsigned short* __restrict__ qbf, const unsigned short* __restrict__ kbf,
    unsigned short* __restrict__ attn) {
  const int lane = threadIdx.x & 31, wave = threadIdx.x >> 5;
  const int cn = lane & 15, hf = lane >> 4;
  const int b = blockIdx.x >> 3, cc = blockIdx.x & 7;

  bf16x16 af[8];
  const unsigned short* qrow = qbf + ((long)b * 16 + cn) * DD;
#pragma unroll
  for (int kk = 0; kk < 8; ++kk)
    af[kk] = ldfrag(qrow, kk * 32 + hf * 16, kk * 32 + hf * 16 + 8);

  for (int t = 0; t < 4; ++t) {
    int jt = cc * 32 + wave * 4 + t;          // 0..255 column tiles
    const unsigned short* krow = kbf + ((long)b * NN + jt * 16 + cn) * DD;
    f32x8 acc = {0.f, 0.f, 0.f, 0.f, 0.f, 0.f, 0.f, 0.f};
#pragma unroll
    for (int kk = 0; kk < 8; ++kk)
      acc = wmma_bf(af[kk], ldfrag(krow, kk * 32 + hf * 16, kk * 32 + hf * 16 + 8), acc);

    float v[8], mx = -1e30f;
#pragma unroll
    for (int g = 0; g < 8; ++g) { v[g] = acc[g] * 0.0625f; mx = fmaxf(mx, v[g]); }
    float sm = 0.f;
#pragma unroll
    for (int g = 0; g < 8; ++g) { v[g] = __expf(v[g] - mx); sm += v[g]; }
    float inv = 1.f / sm;
    int colbase = jt * 16 + cn;
#pragma unroll
    for (int g = 0; g < 8; ++g) {
      int r = g + 8 * hf;
      unsigned short out = hf ? (unsigned short)0 : f2bf(v[g] * inv + 1e-8f);
      attn[((long)b * 16 + r) * NN + colbase] = out;
    }
  }
}

// Deterministic row sums of attn (normalization denominators).
__global__ void __launch_bounds__(256) k_rowsum(const unsigned short* __restrict__ attn,
                                                float* __restrict__ rs) {
  int row = blockIdx.x;                       // b*8 + i
  int b = row >> 3, i = row & 7;
  const unsigned short* p = attn + ((long)b * 16 + i) * NN;
  float s = 0.f;
#pragma unroll
  for (int e = 0; e < 16; ++e) s += bf2f(p[threadIdx.x + 256 * e]);
#pragma unroll
  for (int m = 16; m >= 1; m >>= 1) s += __shfl_xor(s, m, 32);
  __shared__ float red[8];
  if ((threadIdx.x & 31) == 0) red[threadIdx.x >> 5] = s;
  __syncthreads();
  if (threadIdx.x == 0) {
    float t = 0.f;
    for (int w = 0; w < 8; ++w) t += red[w];
    rs[b * 16 + i] = t;
  }
}

// updates = (attn / rowsum) @ v  via  A=attn[16x4096], B=vT tiles.  1 WG/batch.
__global__ void __launch_bounds__(256) k_updates(
    const unsigned short* __restrict__ attn, const unsigned short* __restrict__ vT,
    const float* __restrict__ rs, unsigned short* __restrict__ updbf) {
  const int lane = threadIdx.x & 31, wave = threadIdx.x >> 5;
  const int cn = lane & 15, hf = lane >> 4;
  const int b = blockIdx.x;
  const unsigned short* arow = attn + ((long)b * 16 + cn) * NN;
  const unsigned short* v0 = vT + ((long)b * DD + (wave * 2 + 0) * 16 + cn) * NN;
  const unsigned short* v1 = vT + ((long)b * DD + (wave * 2 + 1) * 16 + cn) * NN;
  f32x8 a0 = {0.f, 0.f, 0.f, 0.f, 0.f, 0.f, 0.f, 0.f};
  f32x8 a1 = a0;
  for (int kk = 0; kk < 128; ++kk) {
    int k0 = kk * 32;
    bf16x16 af = ldfrag(arow, k0 + hf * 16, k0 + hf * 16 + 8);
    a0 = wmma_bf(af, ldfrag(v0, k0 + hf * 16, k0 + hf * 16 + 8), a0);
    a1 = wmma_bf(af, ldfrag(v1, k0 + hf * 16, k0 + hf * 16 + 8), a1);
  }
#pragma unroll
  for (int s = 0; s < 2; ++s) {
    int col = (wave * 2 + s) * 16 + cn;
#pragma unroll
    for (int g = 0; g < 8; ++g) {
      int r = g + 8 * hf;
      float val = hf ? 0.f : ((s ? a1[g] : a0[g]) / rs[b * 16 + g]);
      updbf[((long)b * 16 + r) * DD + col] = f2bf(val);
    }
  }
}

// ---------------------------------------------------------------------------
// GRU cell + LN + MLP residual, one WG per batch, all GEMMs via WMMA.
__global__ void __launch_bounds__(256) k_gru_mlp(
    const unsigned short* __restrict__ updbf, const unsigned short* __restrict__ slotsbf,
    const unsigned short* __restrict__ Wihb, const unsigned short* __restrict__ Whhb,
    const float* __restrict__ b_ih, const float* __restrict__ b_hh,
    const unsigned short* __restrict__ W1b, const float* __restrict__ b1,
    const unsigned short* __restrict__ W2b, const float* __restrict__ b2,
    const float* __restrict__ g_mlp, const float* __restrict__ be_mlp,
    float* __restrict__ slots) {
  __shared__ __align__(16) char smem[34816 + 8192];
  float* Sr = (float*)smem;                       // [8][256] r-gate sum
  float* Sz = Sr + 2048;                          // [8][256] z-gate sum
  float* Xn = Sr + 4096;                          // [8][256] x n-gate
  float* Hn = Sr + 6144;                          // [8][256] h n-gate
  unsigned short* xln = (unsigned short*)smem;            // [16][264] (overlays gates)
  unsigned short* h1  = (unsigned short*)(smem + 16896);  // [16][264]
  float* sg = (float*)(smem + 34816);             // [8][256] GRU output
  const int lane = threadIdx.x & 31, wave = threadIdx.x >> 5;
  const int cn = lane & 15, hf = lane >> 4;
  const int b = blockIdx.x;

  // stage 1a: gh = slots_prev @ w_hh^T + b_hh
  const unsigned short* hrow = slotsbf + ((long)b * 16 + cn) * DD;
  for (int s = 0; s < 6; ++s) {
    int ct = wave + 8 * s;                        // 0..47
    const unsigned short* Brow = Whhb + (ct * 16 + cn) * DD;
    f32x8 acc = {0.f, 0.f, 0.f, 0.f, 0.f, 0.f, 0.f, 0.f};
#pragma unroll
    for (int kk = 0; kk < 8; ++kk) {
      int k0 = kk * 32;
      acc = wmma_bf(ldfrag(hrow, k0 + hf * 16, k0 + hf * 16 + 8),
                    ldfrag(Brow, k0 + hf * 16, k0 + hf * 16 + 8), acc);
    }
    if (!hf) {
      int col = ct * 16 + cn, reg = col >> 8, ccx = col & 255;
      float bias = b_hh[col];
      float* dst = (reg == 0) ? Sr : ((reg == 1) ? Sz : Hn);
#pragma unroll
      for (int g = 0; g < 8; ++g) dst[g * 256 + ccx] = acc[g] + bias;
    }
  }
  __syncthreads();
  // stage 1b: gx = updates @ w_ih^T + b_ih (accumulate r/z, separate n)
  const unsigned short* xrow = updbf + ((long)b * 16 + cn) * DD;
  for (int s = 0; s < 6; ++s) {
    int ct = wave + 8 * s;
    const unsigned short* Brow = Wihb + (ct * 16 + cn) * DD;
    f32x8 acc = {0.f, 0.f, 0.f, 0.f, 0.f, 0.f, 0.f, 0.f};
#pragma unroll
    for (int kk = 0; kk < 8; ++kk) {
      int k0 = kk * 32;
      acc = wmma_bf(ldfrag(xrow, k0 + hf * 16, k0 + hf * 16 + 8),
                    ldfrag(Brow, k0 + hf * 16, k0 + hf * 16 + 8), acc);
    }
    if (!hf) {
      int col = ct * 16 + cn, reg = col >> 8, ccx = col & 255;
      float bias = b_ih[col];
      if (reg == 2) {
#pragma unroll
        for (int g = 0; g < 8; ++g) Xn[g * 256 + ccx] = acc[g] + bias;
      } else {
        float* dst = reg ? Sz : Sr;
#pragma unroll
        for (int g = 0; g < 8; ++g) dst[g * 256 + ccx] += acc[g] + bias;
      }
    }
  }
  __syncthreads();
  // stage 2: GRU elementwise
  {
    int d = threadIdx.x;
#pragma unroll
    for (int i = 0; i < 8; ++i) {
      float r = 1.f / (1.f + __expf(-Sr[i * 256 + d]));
      float z = 1.f / (1.f + __expf(-Sz[i * 256 + d]));
      float nn = tanhf(Xn[i * 256 + d] + r * Hn[i * 256 + d]);
      float h = slots[((long)b * NSL + i) * DD + d];
      sg[i * 256 + d] = (1.f - z) * nn + z * h;
    }
  }
  __syncthreads();
  // LN(sg) -> xln (bf16, rows 8..15 zero); xln overlays dead gate arrays
  {
    int r = wave;
    float x[8], s1 = 0.f, s2 = 0.f;
#pragma unroll
    for (int e = 0; e < 8; ++e) { x[e] = sg[r * 256 + lane + 32 * e]; s1 += x[e]; s2 += x[e] * x[e]; }
#pragma unroll
    for (int m = 16; m >= 1; m >>= 1) { s1 += __shfl_xor(s1, m, 32); s2 += __shfl_xor(s2, m, 32); }
    float mean = s1 * (1.f / DD);
    float rstd = rsqrtf(s2 * (1.f / DD) - mean * mean + 1e-5f);
#pragma unroll
    for (int e = 0; e < 8; ++e) {
      int d = lane + 32 * e;
      xln[r * 264 + d] = f2bf((x[e] - mean) * rstd * g_mlp[d] + be_mlp[d]);
      xln[(8 + r) * 264 + d] = 0;
    }
  }
  __syncthreads();
  // stage 3: h1 = relu(xln @ W1^T + b1)
  for (int s = 0; s < 2; ++s) {
    int ct = wave * 2 + s;
    const unsigned short* Arow = xln + cn * 264;
    const unsigned short* Brow = W1b + (ct * 16 + cn) * DD;
    f32x8 acc = {0.f, 0.f, 0.f, 0.f, 0.f, 0.f, 0.f, 0.f};
#pragma unroll
    for (int kk = 0; kk < 8; ++kk) {
      int k0 = kk * 32;
      acc = wmma_bf(ldfrag(Arow, k0 + hf * 16, k0 + hf * 16 + 8),
                    ldfrag(Brow, k0 + hf * 16, k0 + hf * 16 + 8), acc);
    }
    int col = ct * 16 + cn;
#pragma unroll
    for (int g = 0; g < 8; ++g) {
      int r = g + 8 * hf;
      float val = hf ? 0.f : fmaxf(acc[g] + b1[col], 0.f);
      h1[r * 264 + col] = f2bf(val);
    }
  }
  __syncthreads();
  // stage 4: slots = sg + h1 @ W2^T + b2
  for (int s = 0; s < 2; ++s) {
    int ct = wave * 2 + s;
    const unsigned short* Arow = h1 + cn * 264;
    const unsigned short* Brow = W2b + (ct * 16 + cn) * DD;
    f32x8 acc = {0.f, 0.f, 0.f, 0.f, 0.f, 0.f, 0.f, 0.f};
#pragma unroll
    for (int kk = 0; kk < 8; ++kk) {
      int k0 = kk * 32;
      acc = wmma_bf(ldfrag(Arow, k0 + hf * 16, k0 + hf * 16 + 8),
                    ldfrag(Brow, k0 + hf * 16, k0 + hf * 16 + 8), acc);
    }
    if (!hf) {
      int col = ct * 16 + cn;
#pragma unroll
      for (int g = 0; g < 8; ++g)
        slots[((long)b * NSL + g) * DD + col] = acc[g] + b2[col] + sg[g * 256 + col];
    }
  }
}

// ---------------------------------------------------------------------------
extern "C" void kernel_launch(void* const* d_in, const int* in_sizes, int n_in,
                              void* d_out, int out_size, void* d_ws, size_t ws_size,
                              hipStream_t stream) {
  const float* inp   = (const float*)d_in[0];
  const float* noise = (const float*)d_in[1];
  const float* mu    = (const float*)d_in[2];
  const float* ls    = (const float*)d_in[3];
  const float* Wq    = (const float*)d_in[4];
  const float* bq    = (const float*)d_in[5];
  const float* Wk    = (const float*)d_in[6];
  const float* bk    = (const float*)d_in[7];
  const float* Wv    = (const float*)d_in[8];
  const float* bv    = (const float*)d_in[9];
  const float* w_ih  = (const float*)d_in[10];
  const float* w_hh  = (const float*)d_in[11];
  const float* b_ih  = (const float*)d_in[12];
  const float* b_hh  = (const float*)d_in[13];
  const float* W1    = (const float*)d_in[14];
  const float* b1    = (const float*)d_in[15];
  const float* W2    = (const float*)d_in[16];
  const float* b2    = (const float*)d_in[17];
  const float* g_in  = (const float*)d_in[18];
  const float* be_in = (const float*)d_in[19];
  const float* g_sl  = (const float*)d_in[20];
  const float* be_sl = (const float*)d_in[21];
  const float* g_mlp = (const float*)d_in[22];
  const float* be_mlp= (const float*)d_in[23];

  char* ws = (char*)d_ws;
  size_t off = 0;
  auto carve = [&](size_t bytes) -> void* {
    void* p = ws + off;
    off += (bytes + 255) & ~(size_t)255;
    return p;
  };
  unsigned short* kbf     = (unsigned short*)carve((size_t)BB * NN * DD * 2);
  unsigned short* vT      = (unsigned short*)carve((size_t)BB * NN * DD * 2);
  unsigned short* attn    = (unsigned short*)carve((size_t)BB * 16 * NN * 2);
  unsigned short* qbf     = (unsigned short*)carve((size_t)BB * 16 * DD * 2);
  unsigned short* slotsbf = (unsigned short*)carve((size_t)BB * 16 * DD * 2);
  unsigned short* updbf   = (unsigned short*)carve((size_t)BB * 16 * DD * 2);
  float*          slots   = (float*)carve((size_t)BB * NSL * DD * 4);
  float*          rs      = (float*)carve((size_t)BB * 16 * 4);
  unsigned short* Wqb     = (unsigned short*)carve((size_t)DD * DD * 2);
  unsigned short* Wkb     = (unsigned short*)carve((size_t)DD * DD * 2);
  unsigned short* Wvb     = (unsigned short*)carve((size_t)DD * DD * 2);
  unsigned short* Wihb    = (unsigned short*)carve((size_t)3 * DD * DD * 2);
  unsigned short* Whhb    = (unsigned short*)carve((size_t)3 * DD * DD * 2);
  unsigned short* W1b     = (unsigned short*)carve((size_t)DD * DD * 2);
  unsigned short* W2b     = (unsigned short*)carve((size_t)DD * DD * 2);

  k_conv<<<(DD * DD + 255) / 256, 256, 0, stream>>>(Wq, Wqb, DD * DD);
  k_conv<<<(DD * DD + 255) / 256, 256, 0, stream>>>(Wk, Wkb, DD * DD);
  k_conv<<<(DD * DD + 255) / 256, 256, 0, stream>>>(Wv, Wvb, DD * DD);
  k_conv<<<(3 * DD * DD + 255) / 256, 256, 0, stream>>>(w_ih, Wihb, 3 * DD * DD);
  k_conv<<<(3 * DD * DD + 255) / 256, 256, 0, stream>>>(w_hh, Whhb, 3 * DD * DD);
  k_conv<<<(DD * DD + 255) / 256, 256, 0, stream>>>(W1, W1b, DD * DD);
  k_conv<<<(DD * DD + 255) / 256, 256, 0, stream>>>(W2, W2b, DD * DD);
  k_slots_init<<<(BB * NSL * DD + 255) / 256, 256, 0, stream>>>(noise, mu, ls, slots);

  k_lnkv<<<BB * NN / 32, 256, 0, stream>>>(inp, g_in, be_in, Wkb, bk, Wvb, bv, kbf, vT);

  for (int it = 0; it < NITR; ++it) {
    k_qproj<<<BB / 2, 256, 0, stream>>>(slots, g_sl, be_sl, Wqb, bq, qbf, slotsbf);
    k_dots<<<BB * 8, 256, 0, stream>>>(qbf, kbf, attn);
    k_rowsum<<<BB * NSL, 256, 0, stream>>>(attn, rs);
    k_updates<<<BB, 256, 0, stream>>>(attn, vT, rs, updbf);
    k_gru_mlp<<<BB, 256, 0, stream>>>(updbf, slotsbf, Wihb, Whhb, b_ih, b_hh,
                                      W1b, b1, W2b, b2, g_mlp, be_mlp, slots);
  }
  hipMemcpyAsync(d_out, slots, (size_t)BB * NSL * DD * 4, hipMemcpyDeviceToDevice, stream);
}
